// MambaBlock_90537910600118
// MI455X (gfx1250) — compile-verified, matrix operand order checked by reading, not run on hardware
//
#include <hip/hip_runtime.h>

typedef __attribute__((ext_vector_type(16))) _Float16 v16h;
typedef __attribute__((ext_vector_type(8)))  _Float16 v8h;
typedef __attribute__((ext_vector_type(8)))  float    v8f;

#define HIDN  1024
#define INNER 2048
#define CONVW 4
#define LDIM  16
#define DRNK  64
#define SEQ   2048
#define NXP   96       // DR + 2*L
#define SCH   64       // scan chunk length
#define NCH   (SEQ / SCH)   // 32 chunks

// ---------- CDNA5 async global->LDS helpers (ASYNCcnt-tracked) ----------
// LDS byte address = low 32 bits of the generic (shared-aperture) address.
__device__ __forceinline__ void async_g2lds_b32(void* lds_ptr, const void* gptr) {
  unsigned lds_addr = (unsigned)(size_t)lds_ptr;
  asm volatile("global_load_async_to_lds_b32 %0, %1, off"
               :: "v"(lds_addr), "v"(gptr) : "memory");
}
__device__ __forceinline__ void wait_async0() {
  asm volatile("s_wait_asynccnt 0x0" ::: "memory");
}

// ---------- elementwise conversion helpers ----------
__global__ void k_f32_to_f16(const float* __restrict__ src, _Float16* __restrict__ dst, int n) {
  int i = blockIdx.x * blockDim.x + threadIdx.x;
  if (i < n) dst[i] = (_Float16)src[i];
}

// src is R x C row-major fp32; dst is C x R row-major f16 (transposed)
__global__ void k_transpose_f32_to_f16(const float* __restrict__ src, _Float16* __restrict__ dst,
                                       int R, int C) {
  int i = blockIdx.x * blockDim.x + threadIdx.x;
  if (i < R * C) {
    int r = i / C, c = i % C;
    dst[(size_t)c * R + r] = (_Float16)src[i];
  }
}

// ---------- WMMA GEMM: C[M,N] = A[M,K] x Bt[N,K]^T (f16 in, f32 out) ----------
// One wave computes a 16 x (16*NT) strip: A fragment reused across NT B tiles.
template<int NT>
__global__ __launch_bounds__(32)
void k_wmma_gemm(const _Float16* __restrict__ A,
                 const _Float16* __restrict__ Bt,
                 float* __restrict__ C,
                 int M, int N, int K) {
  const int nt0  = blockIdx.x * NT;   // first tile col of this wave's strip
  const int mt   = blockIdx.y;        // tile row
  const int lane = threadIdx.x;       // 0..31, full wave (EXEC all 1s for WMMA)
  const int r    = lane & 15;         // A-row / B-col within tile
  const int h    = lane >> 4;         // lane half selects K sub-block

  const _Float16* ap = A + (size_t)(mt * 16 + r) * K + h * 8;
  const _Float16* bp[NT];
#pragma unroll
  for (int t = 0; t < NT; ++t)
    bp[t] = Bt + (size_t)((nt0 + t) * 16 + r) * K + h * 8;

  v8f acc[NT];
#pragma unroll
  for (int t = 0; t < NT; ++t) acc[t] = (v8f){};

  for (int k0 = 0; k0 < K; k0 += 32) {
    // keep streaming panels warm: prefetch a few K-steps ahead (global_prefetch_b8)
    if (k0 + 128 < K) {
      __builtin_prefetch(ap + k0 + 128, 0, 1);
      __builtin_prefetch(bp[0] + k0 + 128, 0, 1);
    }
    // 16-bit fragment layout: VGPR0-3 = K[0..7 | 8..15 by lane half],
    // VGPR4-7 = K[16..23 | 24..31 by lane half] -> two contiguous 8-half loads.
    v8h a0 = *(const v8h*)(ap + k0);
    v8h a1 = *(const v8h*)(ap + k0 + 16);
    v16h a;
#pragma unroll
    for (int e = 0; e < 8; ++e) { a[e] = a0[e]; a[e + 8] = a1[e]; }
#pragma unroll
    for (int t = 0; t < NT; ++t) {
      v8h b0 = *(const v8h*)(bp[t] + k0);
      v8h b1 = *(const v8h*)(bp[t] + k0 + 16);
      v16h b;
#pragma unroll
      for (int e = 0; e < 8; ++e) { b[e] = b0[e]; b[e + 8] = b1[e]; }
      acc[t] = __builtin_amdgcn_wmma_f32_16x16x32_f16(false, a, false, b, (short)0, acc[t],
                                                      false, false);
    }
  }
  // C/D layout: lane -> N = lane%16 ; VGPR v -> M = v + 8*(lane/16)
#pragma unroll
  for (int t = 0; t < NT; ++t) {
    float* cp = C + (size_t)(mt * 16 + h * 8) * N + (nt0 + t) * 16 + r;
#pragma unroll
    for (int v = 0; v < 8; ++v) cp[(size_t)v * N] = acc[t][v];
  }
}

// ---------- causal depthwise conv (CONV=4) + bias + SiLU ----------
__global__ void k_conv_silu(const float* __restrict__ xz,      // [S, 2*INNER], xi = cols [0,INNER)
                            const float* __restrict__ conv_w,  // [4,1,INNER]
                            const float* __restrict__ conv_b,  // [INNER]
                            float* __restrict__ xc,            // [S, INNER] f32
                            _Float16* __restrict__ xch) {      // [S, INNER] f16
  int i = blockIdx.x * blockDim.x + threadIdx.x;
  if (i >= SEQ * INNER) return;
  int s = i / INNER, d = i % INNER;
  float acc = conv_b[d];
#pragma unroll
  for (int j = 0; j < CONVW; ++j) {
    int sp = s + j - (CONVW - 1);
    if (sp >= 0) acc += conv_w[j * INNER + d] * xz[(size_t)sp * (2 * INNER) + d];
  }
  float v = acc / (1.0f + __expf(-acc));   // silu
  xc[i]  = v;
  xch[i] = (_Float16)v;
}

// ---------- slice xp[:, :DR] to f16 for the delta GEMM ----------
__global__ void k_extract_delta_f16(const float* __restrict__ xp, _Float16* __restrict__ dxp) {
  int i = blockIdx.x * blockDim.x + threadIdx.x;
  if (i < SEQ * DRNK) {
    int s = i / DRNK, k = i % DRNK;
    dxp[i] = (_Float16)xp[(size_t)s * NXP + k];
  }
}

// ---------- softplus(x + b_delta) in place ----------
__global__ void k_softplus(float* __restrict__ delta, const float* __restrict__ b_delta) {
  int i = blockIdx.x * blockDim.x + threadIdx.x;
  if (i < SEQ * INNER) {
    int d = i % INNER;
    float x = delta[i] + b_delta[d];
    delta[i] = (x > 20.0f) ? x : log1pf(__expf(x));
  }
}

// ============== blocked selective scan (3 passes) ==============
// h[s] = exp(delta*A)*h[s-1] + delta*B*x  is linear, so:
//  pass1: per chunk (64 steps) compute end-state H and chunk A-product
//         (prod exp(dl*a) == exp(a * sum dl) -> one transcendental per chunk)
//  pass2: 32-step serial carry across chunks per (d,l)
//  pass3: re-scan each chunk from its carry-in, fused epilogue -> y (f16)
// Serial depth drops 2048 -> ~64+32; 4096 blocks of parallel chunk work.
// Staging uses CDNA5 async global->LDS copies (no VGPR round-trip).

__global__ __launch_bounds__(256)
void k_scan_pass1(const float* __restrict__ delta,   // [S, INNER]
                  const float* __restrict__ xp,      // [S, NXP]; Bm = cols [64,80)
                  const float* __restrict__ xc,      // [S, INNER]
                  const float* __restrict__ A_log,   // [INNER, L]
                  float* __restrict__ chunkA,        // [NCH, INNER, L]
                  float* __restrict__ chunkH) {      // [NCH, INNER, L]
  __shared__ float sDelta[SCH][16];
  __shared__ float sXc[SCH][16];
  __shared__ float sBm[SCH][16];

  const int tid = threadIdx.x;
  const int p   = tid >> 4;                 // local channel 0..15
  const int l   = tid & 15;                 // state index 0..15
  const int d   = blockIdx.x * 16 + p;
  const int ch  = blockIdx.y;
  const int s0  = ch * SCH;

  const float a = -__expf(A_log[(size_t)d * LDIM + l]);

  // async staging: 1024 elements per array, 4 per thread, no divergence
  for (int t = tid; t < SCH * 16; t += 256) {
    int ss = t >> 4, c = t & 15;
    int s  = s0 + ss;
    int dg = blockIdx.x * 16 + c;
    async_g2lds_b32(&sDelta[ss][c], &delta[(size_t)s * INNER + dg]);
    async_g2lds_b32(&sXc[ss][c],    &xc[(size_t)s * INNER + dg]);
    async_g2lds_b32(&sBm[ss][c],    &xp[(size_t)s * NXP + DRNK + c]);
  }
  wait_async0();
  __syncthreads();

  float h = 0.0f, dsum = 0.0f;
#pragma unroll 4
  for (int ss = 0; ss < SCH; ++ss) {
    float dl  = sDelta[ss][p];
    float xcv = sXc[ss][p];
    float bm  = sBm[ss][l];
    h = __expf(dl * a) * h + dl * bm * xcv;
    dsum += dl;
  }
  size_t idx = ((size_t)ch * INNER + d) * LDIM + l;
  chunkA[idx] = __expf(a * dsum);
  chunkH[idx] = h;
}

__global__ void k_scan_carry(const float* __restrict__ chunkA,
                             const float* __restrict__ chunkH,
                             float* __restrict__ carryIn) {   // [NCH, INNER, L]
  int i = blockIdx.x * blockDim.x + threadIdx.x;   // (d,l) pair
  if (i >= INNER * LDIM) return;
  float c = 0.0f;
  for (int ch = 0; ch < NCH; ++ch) {
    size_t idx = (size_t)ch * (INNER * LDIM) + i;
    carryIn[idx] = c;
    c = chunkA[idx] * c + chunkH[idx];
  }
}

__global__ __launch_bounds__(256)
void k_scan_pass2(const float* __restrict__ delta,
                  const float* __restrict__ xp,      // Bm = cols [64,80), Cm = [80,96)
                  const float* __restrict__ xc,
                  const float* __restrict__ xz,      // z = cols [INNER, 2*INNER)
                  const float* __restrict__ A_log,
                  const float* __restrict__ Dv,
                  const float* __restrict__ carryIn,
                  _Float16* __restrict__ y_h) {      // [S, INNER] f16 -> out-proj GEMM
  __shared__ float sDelta[SCH][16];
  __shared__ float sXc[SCH][16];
  __shared__ float sZ[SCH][16];
  __shared__ float sBm[SCH][16];
  __shared__ float sCm[SCH][16];

  const int tid = threadIdx.x;
  const int p   = tid >> 4;
  const int l   = tid & 15;
  const int d   = blockIdx.x * 16 + p;
  const int ch  = blockIdx.y;
  const int s0  = ch * SCH;

  const float a  = -__expf(A_log[(size_t)d * LDIM + l]);
  const float Dd = Dv[d];

  for (int t = tid; t < SCH * 16; t += 256) {
    int ss = t >> 4, c = t & 15;
    int s  = s0 + ss;
    int dg = blockIdx.x * 16 + c;
    async_g2lds_b32(&sDelta[ss][c], &delta[(size_t)s * INNER + dg]);
    async_g2lds_b32(&sXc[ss][c],    &xc[(size_t)s * INNER + dg]);
    async_g2lds_b32(&sZ[ss][c],     &xz[(size_t)s * (2 * INNER) + INNER + dg]);
    async_g2lds_b32(&sBm[ss][c],    &xp[(size_t)s * NXP + DRNK + c]);
    async_g2lds_b32(&sCm[ss][c],    &xp[(size_t)s * NXP + DRNK + LDIM + c]);
  }
  // overlap: fetch carry-in while async copies are in flight
  float h = carryIn[((size_t)ch * INNER + d) * LDIM + l];
  wait_async0();
  __syncthreads();

  for (int ss = 0; ss < SCH; ++ss) {
    float dl  = sDelta[ss][p];
    float xcv = sXc[ss][p];
    float bm  = sBm[ss][l];
    float cm  = sCm[ss][l];
    h = __expf(dl * a) * h + dl * bm * xcv;          // h = dA*h + dBx
    float contrib = h * cm;
#pragma unroll
    for (int off = 8; off >= 1; off >>= 1)           // reduce over 16 states
      contrib += __shfl_xor(contrib, off, 16);
    if (l == 0) {
      float z = sZ[ss][p];
      float y = (contrib + xcv * Dd) * (z / (1.0f + __expf(-z)));  // +x*D, *silu(z)
      y_h[(size_t)(s0 + ss) * INNER + d] = (_Float16)y;
    }
  }
}

extern "C" void kernel_launch(void* const* d_in, const int* in_sizes, int n_in,
                              void* d_out, int out_size, void* d_ws, size_t ws_size,
                              hipStream_t stream) {
  (void)in_sizes; (void)n_in; (void)out_size; (void)ws_size;
  const float* x       = (const float*)d_in[0];
  const float* W_in    = (const float*)d_in[1];
  const float* conv_w  = (const float*)d_in[2];
  const float* conv_b  = (const float*)d_in[3];
  const float* W_x     = (const float*)d_in[4];
  const float* W_delta = (const float*)d_in[5];
  const float* b_delta = (const float*)d_in[6];
  const float* A_log   = (const float*)d_in[7];
  const float* Dv      = (const float*)d_in[8];
  const float* W_out   = (const float*)d_in[9];
  float* out = (float*)d_out;

  char* ws = (char*)d_ws;
  size_t off = 0;
  auto alloc = [&](size_t bytes) -> void* {
    void* p = ws + off;
    off += (bytes + 255) & ~(size_t)255;
    return p;
  };

  _Float16* xh     = (_Float16*)alloc((size_t)SEQ * HIDN * 2);
  _Float16* Wint   = (_Float16*)alloc((size_t)2 * INNER * HIDN * 2);   // W_in^T  [4096,1024]
  float*    xz     = (float*)   alloc((size_t)SEQ * 2 * INNER * 4);
  float*    xc     = (float*)   alloc((size_t)SEQ * INNER * 4);
  _Float16* xch    = (_Float16*)alloc((size_t)SEQ * INNER * 2);
  _Float16* Wxt    = (_Float16*)alloc((size_t)NXP * INNER * 2);        // W_x^T   [96,2048]
  float*    xp     = (float*)   alloc((size_t)SEQ * NXP * 4);
  _Float16* dxph   = (_Float16*)alloc((size_t)SEQ * DRNK * 2);
  _Float16* Wdt    = (_Float16*)alloc((size_t)INNER * DRNK * 2);       // W_delta^T [2048,64]
  float*    delta  = (float*)   alloc((size_t)SEQ * INNER * 4);
  _Float16* yh     = (_Float16*)alloc((size_t)SEQ * INNER * 2);
  _Float16* Wot    = (_Float16*)alloc((size_t)HIDN * INNER * 2);       // W_out^T [1024,2048]
  float*    chunkA = (float*)   alloc((size_t)NCH * INNER * LDIM * 4);
  float*    chunkH = (float*)   alloc((size_t)NCH * INNER * LDIM * 4);
  float*    carry  = (float*)   alloc((size_t)NCH * INNER * LDIM * 4);

  const int T = 256;
  auto blocks = [](long n, int t) { return (int)((n + t - 1) / t); };

  // 1) in-proj GEMM: xz = x @ W_in  (2048x1024x4096)
  k_f32_to_f16<<<blocks((long)SEQ * HIDN, T), T, 0, stream>>>(x, xh, SEQ * HIDN);
  k_transpose_f32_to_f16<<<blocks((long)HIDN * 2 * INNER, T), T, 0, stream>>>(W_in, Wint, HIDN, 2 * INNER);
  k_wmma_gemm<4><<<dim3(2 * INNER / 64, SEQ / 16), 32, 0, stream>>>(xh, Wint, xz, SEQ, 2 * INNER, HIDN);

  // 2) causal depthwise conv + SiLU
  k_conv_silu<<<blocks((long)SEQ * INNER, T), T, 0, stream>>>(xz, conv_w, conv_b, xc, xch);

  // 3) x-proj GEMM: xp = xc @ W_x  (N = 96)
  k_transpose_f32_to_f16<<<blocks((long)INNER * NXP, T), T, 0, stream>>>(W_x, Wxt, INNER, NXP);
  k_wmma_gemm<2><<<dim3(NXP / 32, SEQ / 16), 32, 0, stream>>>(xch, Wxt, xp, SEQ, NXP, INNER);

  // 4) delta GEMM + softplus
  k_extract_delta_f16<<<blocks((long)SEQ * DRNK, T), T, 0, stream>>>(xp, dxph);
  k_transpose_f32_to_f16<<<blocks((long)DRNK * INNER, T), T, 0, stream>>>(W_delta, Wdt, DRNK, INNER);
  k_wmma_gemm<4><<<dim3(INNER / 64, SEQ / 16), 32, 0, stream>>>(dxph, Wdt, delta, SEQ, INNER, DRNK);
  k_softplus<<<blocks((long)SEQ * INNER, T), T, 0, stream>>>(delta, b_delta);

  // 5) blocked selective scan + gating -> yh (f16)
  k_scan_pass1<<<dim3(INNER / 16, NCH), 256, 0, stream>>>(delta, xp, xc, A_log, chunkA, chunkH);
  k_scan_carry<<<blocks((long)INNER * LDIM, T), T, 0, stream>>>(chunkA, chunkH, carry);
  k_scan_pass2<<<dim3(INNER / 16, NCH), 256, 0, stream>>>(delta, xp, xc, xz, A_log, Dv, carry, yh);

  // 6) out-proj GEMM: out = y @ W_out
  k_transpose_f32_to_f16<<<blocks((long)INNER * HIDN, T), T, 0, stream>>>(W_out, Wot, INNER, HIDN);
  k_wmma_gemm<4><<<dim3(HIDN / 64, SEQ / 16), 32, 0, stream>>>(yh, Wot, out, SEQ, HIDN, INNER);
}
